// GraphormerMultiheadAttention_15444702396548
// MI455X (gfx1250) — compile-verified, matrix operand order checked by reading, not run on hardware
//
#include <hip/hip_runtime.h>
#include <hip/hip_bf16.h>

// ---------------- problem constants ----------------
constexpr int Bn = 4;
constexpr int Sn = 2048;
constexpr int En = 512;
constexpr int Hn = 8;
constexpr int Dn = 64;        // E / H
constexpr int BSn = Bn * Sn;  // 8192 rows
constexpr int NCn = Sn / 64;  // key chunks of 64

typedef __bf16 bf16;
typedef __attribute__((ext_vector_type(16))) __bf16 v16bf_t;
typedef __attribute__((ext_vector_type(8)))  float  v8f_t;

#define NEG_INF (-__builtin_inff())

// Load 16 bf16 elements in the WMMA A/B operand layout for one 16x16x32 step.
// lane<16 : row = base_row + lane,     K chunks [kk32+0..7],  [kk32+16..23]
// lane>=16: row = base_row + lane-16,  K chunks [kk32+8..15], [kk32+24..31]
__device__ __forceinline__ v16bf_t load16(const bf16* base, int row, int rowstride,
                                          int kk32, int lane) {
    const int half = lane >> 4;
    const bf16* p = base + (size_t)(row + (lane & 15)) * rowstride + kk32 + half * 8;
    union { v16bf_t v; uint4 u[2]; } t;
    t.u[0] = *(const uint4*)(p);
    t.u[1] = *(const uint4*)(p + 16);
    return t.v;
}

__device__ __forceinline__ v8f_t wmma_bf16(v16bf_t a, v16bf_t b, v8f_t c) {
    return __builtin_amdgcn_wmma_f32_16x16x32_bf16(false, a, false, b, (short)0, c,
                                                   false, false);
}

// CDNA5 async copy: 16 bytes global -> LDS, tracked by ASYNCcnt.
// lds_off = wave-relative LDS byte address (low 32 bits of a shared pointer).
__device__ __forceinline__ void async_ld_b128(unsigned lds_off, const void* gptr) {
    asm volatile("global_load_async_to_lds_b128 %0, %1, off"
                 :: "v"(lds_off), "v"((unsigned long long)(size_t)gptr)
                 : "memory");
}
__device__ __forceinline__ void wait_async0() {
    asm volatile("s_wait_asynccnt 0" ::: "memory");
}
__device__ __forceinline__ unsigned lds_off32(const void* p) {
    return (unsigned)(unsigned long long)(size_t)p;
}

// ---------------- stage 0: fp32 -> bf16 convert ----------------
__global__ void cvt_bf16_kernel(const float* __restrict__ src, bf16* __restrict__ dst,
                                int n4) {
    int i = blockIdx.x * 256 + threadIdx.x;
    if (i < n4) {
        float4 f = ((const float4*)src)[i];
        union { unsigned long long u; bf16 e[4]; } pk;
        pk.e[0] = (bf16)f.x; pk.e[1] = (bf16)f.y;
        pk.e[2] = (bf16)f.z; pk.e[3] = (bf16)f.w;
        ((unsigned long long*)dst)[i] = pk.u;
    }
}

// ---------------- stage 1: fused QKV projection ----------------
// Y[m,n] = sum_e Xb[m,e] * W[n,e] (+bias).  z=0:Q (scaled 1/8, [B,H,S,D])
// z=1:K ([B,H,S,D])   z=2:V (stored transposed [B,H,D,S])
__global__ __launch_bounds__(256) void proj_qkv_kernel(
    const bf16* __restrict__ Xb,
    const bf16* __restrict__ Wqb, const bf16* __restrict__ Wkb,
    const bf16* __restrict__ Wvb,
    const float* __restrict__ bq, const float* __restrict__ bk,
    const float* __restrict__ bv,
    bf16* __restrict__ Qg, bf16* __restrict__ Kg, bf16* __restrict__ Vt) {
    const int lane = threadIdx.x & 31;
    const int wave = threadIdx.x >> 5;
    const int half = lane >> 4;
    const int l15  = lane & 15;
    const int m0 = blockIdx.x * 128 + wave * 16;   // 16 output rows per wave
    const int n0 = blockIdx.y * 64;                // 64 output cols per wave
    const int z  = blockIdx.z;

    const bf16*  Wb   = (z == 0) ? Wqb : (z == 1) ? Wkb : Wvb;
    const float* bias = (z == 0) ? bq  : (z == 1) ? bk  : bv;

    v8f_t acc[4] = {};
    for (int ks = 0; ks < En / 32; ++ks) {
        v16bf_t a = load16(Xb, m0, En, ks * 32, lane);
        #pragma unroll
        for (int nt = 0; nt < 4; ++nt) {
            v16bf_t b = load16(Wb, n0 + nt * 16, En, ks * 32, lane);
            acc[nt] = wmma_bf16(a, b, acc[nt]);
        }
    }

    const float scale = (z == 0) ? 0.125f : 1.0f;  // fold 1/sqrt(D) into Q
    const int mrow0 = m0 + half * 8;               // first C row owned by this lane
    const int bb = mrow0 / Sn;                     // batch (tile never crosses batch)
    const int s0 = mrow0 % Sn;

    #pragma unroll
    for (int nt = 0; nt < 4; ++nt) {
        const int n  = n0 + nt * 16 + l15;
        const int hh = n >> 6;
        const int d  = n & 63;
        const float bi = bias[n];
        if (z == 2) {
            // V transposed: consecutive C rows -> consecutive s -> packed store
            union { uint4 u; bf16 e[8]; } pk;
            #pragma unroll
            for (int r = 0; r < 8; ++r) pk.e[r] = (bf16)(acc[nt][r] + bi);
            bf16* dst = Vt + ((size_t)(bb * Hn + hh) * Dn + d) * Sn + s0;
            *(uint4*)dst = pk.u;
        } else {
            bf16* dst = ((z == 0) ? Qg : Kg) +
                        ((size_t)(bb * Hn + hh) * Sn + s0) * Dn + d;
            #pragma unroll
            for (int r = 0; r < 8; ++r)
                dst[(size_t)r * Dn] = (bf16)((acc[nt][r] + bi) * scale);
        }
    }
}

// ---------------- stage 2: flash attention ----------------
// S^T = K*Q^T (per-lane = one query column), bias+mask, online softmax,
// O^T = V^T * P^T (P^T's C-layout IS the B-operand layout after bf16 packing).
// K/V^T chunks are double-buffered in LDS via async global->LDS copies so the
// next chunk's HBM fetch overlaps the current chunk's WMMA/softmax work.
__global__ __launch_bounds__(256) void attn_kernel(
    const bf16* __restrict__ Qg, const bf16* __restrict__ Kg,
    const bf16* __restrict__ Vt,
    const float* __restrict__ eb, const int* __restrict__ msk,
    bf16* __restrict__ AO) {
    __shared__ bf16 sK[2][64 * 64];   // [buf][key][d]   2 x 8KB
    __shared__ bf16 sV[2][64 * 64];   // [buf][d][key]   2 x 8KB

    const int lane = threadIdx.x & 31;
    const int wave = threadIdx.x >> 5;
    const int half = lane >> 4;
    const int l15  = lane & 15;
    const int bh = blockIdx.x >> 4;       // 0..31 = b*H+h
    const int qt = blockIdx.x & 15;       // query tile of 128
    const int b  = bh >> 3;
    const int h  = bh & 7;
    const int qrow0 = qt * 128 + wave * 16;    // 16 queries per wave
    const int qi    = qrow0 + l15;             // this lane's query (fixed!)

    const bf16* kgbase = Kg + (size_t)bh * Sn * Dn;   // [key][d]
    const bf16* vgbase = Vt + (size_t)bh * Dn * Sn;   // [d][key]

    // per-thread staging addresses: 2 x 16B chunks for K and for V^T
    const int idx0 = threadIdx.x;          // chunk ids 0..511 (idx0, idx0+256)
    auto stage_async = [&](int kc, int buf) {
        #pragma unroll
        for (int i = 0; i < 2; ++i) {
            const int idx = idx0 + i * 256;
            const int row = idx >> 3;            // 0..63
            const int c8  = (idx & 7) * 8;       // 0,8,..,56
            async_ld_b128(lds_off32(&sK[buf][idx * 8]),
                          kgbase + ((size_t)(kc * 64 + row)) * Dn + c8);
            async_ld_b128(lds_off32(&sV[buf][idx * 8]),
                          vgbase + (size_t)row * Sn + kc * 64 + c8);
        }
    };

    // Q as B-operand (lane = query column, contiguous d per lane)
    v16bf_t qreg[2];
    #pragma unroll
    for (int kk = 0; kk < 2; ++kk)
        qreg[kk] = load16(Qg + (size_t)bh * Sn * Dn, qrow0, Dn, kk * 32, lane);

    v8f_t ot[4] = {};
    float mprev = NEG_INF;
    float lsum  = 0.0f;

    const float* ebrow = eb  + ((size_t)b * Sn + qi) * Sn;
    const int*   mrow  = msk + ((size_t)b * Sn + qi) * Sn;

    // prologue: stage chunk 0
    stage_async(0, 0);
    wait_async0();
    __syncthreads();

    for (int kc = 0; kc < NCn; ++kc) {
        const int cur = kc & 1;
        if (kc + 1 < NCn) {
            stage_async(kc + 1, cur ^ 1);            // overlap with compute
            __builtin_prefetch(ebrow + (kc + 1) * 64, 0, 1);   // next bias row chunk
            __builtin_prefetch(mrow  + (kc + 1) * 64, 0, 1);   // next mask row chunk
        }
        const bf16* cK = sK[cur];
        const bf16* cV = sV[cur];

        // S^T tile: [64 keys x 16 queries], 4 key-subtiles x 2 d-steps
        v8f_t st[4] = {};
        #pragma unroll
        for (int kk = 0; kk < 2; ++kk) {
            #pragma unroll
            for (int mt = 0; mt < 4; ++mt) {
                v16bf_t a = load16(cK, mt * 16, 64, kk * 32, lane);
                st[mt] = wmma_bf16(a, qreg[kk], st[mt]);
            }
        }

        // edge bias + mask (vector loads, contiguous along k for this lane's q)
        #pragma unroll
        for (int mt = 0; mt < 4; ++mt) {
            const int kb = kc * 64 + mt * 16 + half * 8;
            const float4 e0 = *(const float4*)(ebrow + kb);
            const float4 e1 = *(const float4*)(ebrow + kb + 4);
            const int4   g0 = *(const int4*)(mrow + kb);
            const int4   g1 = *(const int4*)(mrow + kb + 4);
            const float ev[8] = {e0.x, e0.y, e0.z, e0.w, e1.x, e1.y, e1.z, e1.w};
            const int   mv[8] = {g0.x, g0.y, g0.z, g0.w, g1.x, g1.y, g1.z, g1.w};
            #pragma unroll
            for (int r = 0; r < 8; ++r) {
                float s = st[mt][r] + ev[r];
                st[mt][r] = (mv[r] == 0) ? NEG_INF : s;
            }
        }

        // online softmax (fixed q per lane; combine the two 16-lane halves once)
        float mc = mprev;
        #pragma unroll
        for (int mt = 0; mt < 4; ++mt)
            #pragma unroll
            for (int r = 0; r < 8; ++r) mc = fmaxf(mc, st[mt][r]);
        mc = fmaxf(mc, __shfl_xor(mc, 16, 32));

        const float rescale = __expf(mprev - mc);
        lsum *= rescale;
        #pragma unroll
        for (int dt = 0; dt < 4; ++dt)
            #pragma unroll
            for (int r = 0; r < 8; ++r) ot[dt][r] *= rescale;

        // P^T -> bf16 B-operand registers (pairwise pack, no lane exchange)
        union { v16bf_t v; bf16 e[16]; } pb[2];
        #pragma unroll
        for (int kk = 0; kk < 2; ++kk) {
            #pragma unroll
            for (int r = 0; r < 8; ++r) {
                float p0 = __expf(st[2 * kk][r] - mc);
                float p1 = __expf(st[2 * kk + 1][r] - mc);
                lsum += p0 + p1;
                pb[kk].e[r]     = (bf16)p0;   // keys kk*32 + half*8 + r
                pb[kk].e[8 + r] = (bf16)p1;   // keys kk*32 + 16 + half*8 + r
            }
        }
        mprev = mc;

        // O^T += V^T * P^T   (A = V^T rows from LDS, contiguous along keys)
        #pragma unroll
        for (int kk = 0; kk < 2; ++kk) {
            #pragma unroll
            for (int dt = 0; dt < 4; ++dt) {
                v16bf_t a = load16(cV, dt * 16, 64, kk * 32, lane);
                ot[dt] = wmma_bf16(a, pb[kk].v, ot[dt]);
            }
        }

        wait_async0();       // my async writes to the other buffer are done
        __syncthreads();     // everyone done reading cur + writes visible
    }

    // finalize: combine half-lane l sums, normalize, packed store along d
    const float ltot = lsum + __shfl_xor(lsum, 16, 32);
    const float inv  = 1.0f / ltot;
    bf16* aorow = AO + ((size_t)(b * Sn + qi)) * En + h * Dn;
    #pragma unroll
    for (int dt = 0; dt < 4; ++dt) {
        union { uint4 u; bf16 e[8]; } pk;
        #pragma unroll
        for (int r = 0; r < 8; ++r) pk.e[r] = (bf16)(ot[dt][r] * inv);
        *(uint4*)(aorow + dt * 16 + half * 8) = pk.u;
    }
}

// ---------------- stage 3: output projection (fp32 result) ----------------
__global__ __launch_bounds__(256) void proj_out_kernel(
    const bf16* __restrict__ AO, const bf16* __restrict__ Wob,
    const float* __restrict__ bo, float* __restrict__ out) {
    const int lane = threadIdx.x & 31;
    const int wave = threadIdx.x >> 5;
    const int half = lane >> 4;
    const int l15  = lane & 15;
    const int m0 = blockIdx.x * 128 + wave * 16;
    const int n0 = blockIdx.y * 64;

    v8f_t acc[4] = {};
    for (int ks = 0; ks < En / 32; ++ks) {
        v16bf_t a = load16(AO, m0, En, ks * 32, lane);
        #pragma unroll
        for (int nt = 0; nt < 4; ++nt) {
            v16bf_t b = load16(Wob, n0 + nt * 16, En, ks * 32, lane);
            acc[nt] = wmma_bf16(a, b, acc[nt]);
        }
    }
    const int mrow0 = m0 + half * 8;
    #pragma unroll
    for (int nt = 0; nt < 4; ++nt) {
        const int n = n0 + nt * 16 + l15;
        const float bi = bo[n];
        #pragma unroll
        for (int r = 0; r < 8; ++r)
            out[(size_t)(mrow0 + r) * En + n] = acc[nt][r] + bi;
    }
}

// ---------------- host launcher ----------------
extern "C" void kernel_launch(void* const* d_in, const int* in_sizes, int n_in,
                              void* d_out, int out_size, void* d_ws, size_t ws_size,
                              hipStream_t stream) {
    const float* x  = (const float*)d_in[0];
    const float* eb = (const float*)d_in[1];
    const int*   mk = (const int*)d_in[2];
    const float* Wq = (const float*)d_in[3];
    const float* bq = (const float*)d_in[4];
    const float* Wk = (const float*)d_in[5];
    const float* bk = (const float*)d_in[6];
    const float* Wv = (const float*)d_in[7];
    const float* bv = (const float*)d_in[8];
    const float* Wo = (const float*)d_in[9];
    const float* bo = (const float*)d_in[10];
    float* out = (float*)d_out;

    char* ws = (char*)d_ws;
    const size_t MB = 1ull << 20;
    const size_t KB = 1ull << 10;
    bf16* Xb  = (bf16*)(ws + 0);                 // 8 MiB  [8192,512]
    bf16* Wqb = (bf16*)(ws + 8 * MB);            // 512 KiB each
    bf16* Wkb = (bf16*)(ws + 8 * MB + 512 * KB);
    bf16* Wvb = (bf16*)(ws + 9 * MB);
    bf16* Wob = (bf16*)(ws + 9 * MB + 512 * KB);
    bf16* Qg  = (bf16*)(ws + 10 * MB);           // 8 MiB [B,H,S,D] (pre-scaled)
    bf16* Kg  = (bf16*)(ws + 18 * MB);           // 8 MiB [B,H,S,D]
    bf16* Vt  = (bf16*)(ws + 26 * MB);           // 8 MiB [B,H,D,S]
    bf16* AO  = (bf16*)(ws + 34 * MB);           // 8 MiB [B*S, E]

    // stage 0: converts
    cvt_bf16_kernel<<<(BSn * En / 4 + 255) / 256, 256, 0, stream>>>(x, Xb, BSn * En / 4);
    cvt_bf16_kernel<<<(En * En / 4 + 255) / 256, 256, 0, stream>>>(Wq, Wqb, En * En / 4);
    cvt_bf16_kernel<<<(En * En / 4 + 255) / 256, 256, 0, stream>>>(Wk, Wkb, En * En / 4);
    cvt_bf16_kernel<<<(En * En / 4 + 255) / 256, 256, 0, stream>>>(Wv, Wvb, En * En / 4);
    cvt_bf16_kernel<<<(En * En / 4 + 255) / 256, 256, 0, stream>>>(Wo, Wob, En * En / 4);

    // stage 1: QKV projections (z selects Q/K/V)
    proj_qkv_kernel<<<dim3(BSn / 128, En / 64, 3), 256, 0, stream>>>(
        Xb, Wqb, Wkb, Wvb, bq, bk, bv, Qg, Kg, Vt);

    // stage 2: flash attention (one block per (b,h,128 queries))
    attn_kernel<<<dim3(Bn * Hn * (Sn / 128)), 256, 0, stream>>>(Qg, Kg, Vt, eb, mk, AO);

    // stage 3: output projection
    proj_out_kernel<<<dim3(BSn / 128, En / 64), 256, 0, stream>>>(AO, Wob, bo, out);
}
